// MultiHeadedAttention_77438260347067
// MI455X (gfx1250) — matrix operand order repaired and checked by reading, not compiled, not run on hardware
//
#include <hip/hip_runtime.h>
#include <hip/hip_bf16.h>
#include <math.h>

#define NTOK   1024
#define DMODEL 512
#define NHEAD  8
#define DK     64
#define DEDGE  64

typedef __attribute__((ext_vector_type(16))) __bf16 v16bf;
typedef __attribute__((ext_vector_type(8)))  float  v8f;

union BF16Frag {
  unsigned short u[16];
  uint4 q[2];
  v16bf v;
};

// ---------- helpers ----------

__device__ __forceinline__ unsigned short f2bf(float f) {
  union { float f; unsigned int u; } x; x.f = f;
  unsigned int r = x.u + 0x7FFFu + ((x.u >> 16) & 1u);   // RNE
  return (unsigned short)(r >> 16);
}

// A fragment: 16x32 bf16 tile, element (m,k) at p[m*rs + k].
// k order used here: lanes 0-15 (m=lane) hold K=0..15 ; lanes 16-31 hold K=16..31.
// Requires p and rs such that (p + m*rs + kb) is 16B aligned (rs multiple of 8, p 16B aligned).
__device__ __forceinline__ v16bf load_a_frag(const unsigned short* p, int rs) {
  const int lane = threadIdx.x & 31;
  const int m  = lane & 15;
  const int kb = (lane & 16) ? 16 : 0;
  const unsigned short* base = p + m * rs + kb;
  BF16Frag f;
  f.q[0] = *(const uint4*)(base);        // K = kb .. kb+7
  f.q[1] = *(const uint4*)(base + 8);    // K = kb+8 .. kb+15
  return f.v;
}

// Same but rows m >= mmax read as zero (M-padding).
__device__ __forceinline__ v16bf load_a_frag_masked(const unsigned short* p, int rs, int mmax) {
  const int lane = threadIdx.x & 31;
  const int m  = lane & 15;
  const int kb = (lane & 16) ? 16 : 0;
  BF16Frag f;
  if (m < mmax) {
    const unsigned short* base = p + m * rs + kb;
    f.q[0] = *(const uint4*)(base);
    f.q[1] = *(const uint4*)(base + 8);
  } else {
    f.q[0] = make_uint4(0u, 0u, 0u, 0u);
    f.q[1] = make_uint4(0u, 0u, 0u, 0u);
  }
  return f.v;
}

// A fragment from fp32 source (converted on the fly; used only for the tiny 16x64 T matrix).
__device__ __forceinline__ v16bf load_a_frag_f32(const float* p, int rs) {
  const int lane = threadIdx.x & 31;
  const int m  = lane & 15;
  const int kb = (lane & 16) ? 16 : 0;
  BF16Frag f;
#pragma unroll
  for (int e = 0; e < 8; ++e) f.u[e]     = f2bf(p[m * rs + kb + e]);
#pragma unroll
  for (int e = 0; e < 8; ++e) f.u[8 + e] = f2bf(p[m * rs + kb + 8 + e]);
  return f.v;
}

// B fragment from pre-packed (fragment-major) storage: tile = 512 contiguous ushort,
// lane's 16 elements contiguous at packed + tile*512 + lane*16.
__device__ __forceinline__ v16bf load_b_packed(const unsigned short* packed, int tile) {
  const unsigned short* p = packed + (size_t)tile * 512 + (threadIdx.x & 31) * 16;
  BF16Frag f;
  f.q[0] = *(const uint4*)(p);
  f.q[1] = *(const uint4*)(p + 8);
  return f.v;
}

// B fragment from a TRANSPOSED tile: element (k,n) at pT[n*rs + k]  (so per-lane reads
// are contiguous in k). k order used here: lanes 0-15 (n=lane) hold K=0..15 ; lanes 16-31 hold K=16..31.
__device__ __forceinline__ v16bf load_bT_frag(const unsigned short* pT, int rs) {
  const int lane = threadIdx.x & 31;
  const int n  = lane & 15;
  const int kb = (lane & 16) ? 16 : 0;
  const unsigned short* base = pT + n * rs + kb;
  BF16Frag f;
  f.q[0] = *(const uint4*)(base);
  f.q[1] = *(const uint4*)(base + 8);
  return f.v;
}

// Block-diagonal Q operand for the QK term: B(k,n) = (global_k/64==n) ? q[global_k] : 0.
// For k-step ks (k0 = ks*32) only lane n == ks/2 is nonzero, contiguous in q.
__device__ __forceinline__ v16bf load_b_qdiag(const unsigned short* qrow, int ks) {
  const int lane = threadIdx.x & 31;
  const int n  = lane & 15;
  const int kb = (lane & 16) ? 16 : 0;
  BF16Frag f;
  if (n == (ks >> 1)) {
    const unsigned short* base = qrow + ks * 32 + kb;
    f.q[0] = *(const uint4*)(base);
    f.q[1] = *(const uint4*)(base + 8);
  } else {
    f.q[0] = make_uint4(0u, 0u, 0u, 0u);
    f.q[1] = make_uint4(0u, 0u, 0u, 0u);
  }
  return f.v;
}

// Q^T operand for the structure term: B(k,n) = (n<8) ? q[n*64 + k0+k] : 0.
__device__ __forceinline__ v16bf load_b_qt(const unsigned short* qrow, int k0) {
  const int lane = threadIdx.x & 31;
  const int n  = lane & 15;
  const int kb = (lane & 16) ? 16 : 0;
  BF16Frag f;
  if (n < 8) {
    const unsigned short* base = qrow + n * 64 + k0 + kb;
    f.q[0] = *(const uint4*)(base);
    f.q[1] = *(const uint4*)(base + 8);
  } else {
    f.q[0] = make_uint4(0u, 0u, 0u, 0u);
    f.q[1] = make_uint4(0u, 0u, 0u, 0u);
  }
  return f.v;
}

__device__ __forceinline__ v8f wmma_bf16(v16bf a, v16bf b, v8f c) {
  return __builtin_amdgcn_wmma_f32_16x16x32_bf16(false, a, false, b, (short)0, c, false, false);
}

// ---------- kernels ----------

__global__ void cvt_f32_bf16(const float* __restrict__ in, unsigned short* __restrict__ out, int n) {
  for (int i = blockIdx.x * blockDim.x + threadIdx.x; i < n; i += gridDim.x * blockDim.x)
    out[i] = f2bf(in[i]);
}

// Pack fp32 W[K,N] into fragment-major bf16 B layout:
// tile t = (k/32)*(N/16) + n/16 ; within tile: lane = (k&16)+(n&15), elem = k&15.
__global__ void pack_b_bf16(const float* __restrict__ W, unsigned short* __restrict__ out,
                            int K, int N) {
  const int total = K * N;
  const int ntiles = N >> 4;
  for (int idx = blockIdx.x * blockDim.x + threadIdx.x; idx < total;
       idx += gridDim.x * blockDim.x) {
    int tile = idx >> 9;
    int r    = idx & 511;
    int lane = r >> 4;
    int e    = r & 15;
    int ks = tile / ntiles;
    int nt = tile - ks * ntiles;
    int k = ks * 32 + (lane & 16) + e;
    int n = nt * 16 + (lane & 15);
    out[idx] = f2bf(W[(size_t)k * N + n]);
  }
}

// C[M,N] = act(A_bf16[M,K] @ Bpacked_bf16[K,N] + bias)   (block: 16 x 128 tile, 8 waves)
// pack_out: write outBF in fragment-major B layout (for V feeding pv_kernel).
__global__ __launch_bounds__(256) void gemm_bf16_kernel(
    const unsigned short* __restrict__ A, const unsigned short* __restrict__ Bp,
    const float* __restrict__ bias, float* __restrict__ outF,
    unsigned short* __restrict__ outBF, int M, int N, int K, int do_relu, int pack_out) {
  const int wave = threadIdx.x >> 5;
  const int lane = threadIdx.x & 31;
  const int i0 = blockIdx.x * 16;
  const int n0 = blockIdx.y * 128 + wave * 16;
  if (n0 >= N) return;
  const int ntiles = N >> 4;
  const int ntile  = n0 >> 4;
  v8f acc = {};
  for (int ks = 0; ks < (K >> 5); ++ks) {
    v16bf a = load_a_frag(A + (size_t)i0 * K + ks * 32, K);
    v16bf b = load_b_packed(Bp, ks * ntiles + ntile);
    acc = wmma_bf16(a, b, acc);
  }
  const int n  = lane & 15;
  const int mb = (lane & 16) ? 8 : 0;
  const int col = n0 + n;
  const float bb = bias ? bias[col] : 0.f;
  float vals[8];
#pragma unroll
  for (int r = 0; r < 8; ++r) {
    float v = acc[r] + bb;
    vals[r] = do_relu ? fmaxf(v, 0.f) : v;
  }
  if (outF) {
#pragma unroll
    for (int r = 0; r < 8; ++r)
      outF[(size_t)(i0 + mb + r) * N + col] = vals[r];
  }
  if (outBF) {
    if (pack_out) {
#pragma unroll
      for (int r = 0; r < 8; ++r) {
        int kk = i0 + mb + r;
        size_t pidx = ((size_t)(kk >> 5) * ntiles + (col >> 4)) * 512 +
                      (size_t)((kk & 16) + (col & 15)) * 16 + (kk & 15);
        outBF[pidx] = f2bf(vals[r]);
      }
    } else {
#pragma unroll
      for (int r = 0; r < 8; ++r)
        outBF[(size_t)(i0 + mb + r) * N + col] = f2bf(vals[r]);
    }
  }
}

// scores[h,i,j] = ( q_ih . k_jh  +  q_ih . relu(S[i,j]@Wsk+bsk) ) / sqrt(dk)
// grid (8 j-slices, 1024 i), block 256 = 8 waves, each wave owns a 16-j tile.
__global__ __launch_bounds__(256) void scores_kernel(
    const float* __restrict__ structure, const unsigned short* __restrict__ Qbf,
    const unsigned short* __restrict__ Kbf, const unsigned short* __restrict__ Wskp,
    const float* __restrict__ bsk, float* __restrict__ scores) {
  const int i    = blockIdx.y;
  const int wave = threadIdx.x >> 5;
  const int lane = threadIdx.x & 31;
  const int j0   = blockIdx.x * 128 + wave * 16;
  __shared__ __align__(16) unsigned short S_lds [8][16][72];
  __shared__ __align__(16) unsigned short SK_lds[8][16][72];

  // stage structure tile (16 x 64) -> LDS bf16, coalesced
  const float* Srow = structure + ((size_t)i * NTOK + j0) * DEDGE;
  for (int idx = lane; idx < 16 * 64; idx += 32) {
    int r = idx >> 6, c = idx & 63;
    S_lds[wave][r][c] = f2bf(Srow[(size_t)r * DEDGE + c]);
  }
  __syncthreads();

  const int n  = lane & 15;
  const int mb = (lane & 16) ? 8 : 0;

  // SK = relu(S @ Wsk + bsk)  (16 x 64)   Wsk packed: 2 k-tiles x 4 n-tiles
#pragma unroll
  for (int nt = 0; nt < 4; ++nt) {
    v8f acc = {};
#pragma unroll
    for (int ks = 0; ks < 2; ++ks) {
      v16bf a = load_a_frag(&S_lds[wave][0][ks * 32], 72);
      v16bf b = load_b_packed(Wskp, ks * 4 + nt);
      acc = wmma_bf16(a, b, acc);
    }
    int col = nt * 16 + n;
    float bb = bsk[col];
#pragma unroll
    for (int r = 0; r < 8; ++r)
      SK_lds[wave][mb + r][col] = f2bf(fmaxf(acc[r] + bb, 0.f));
  }
  __syncthreads();

  // one accumulator: QK term (K=512, block-diag Q) + structure term (K=64)
  const unsigned short* qrow = Qbf + (size_t)i * DMODEL;
  v8f acc = {};
#pragma unroll
  for (int ks = 0; ks < 16; ++ks) {
    v16bf a = load_a_frag(Kbf + (size_t)j0 * DMODEL + ks * 32, DMODEL);
    v16bf b = load_b_qdiag(qrow, ks);
    acc = wmma_bf16(a, b, acc);
  }
#pragma unroll
  for (int ks = 0; ks < 2; ++ks) {
    v16bf a = load_a_frag(&SK_lds[wave][0][ks * 32], 72);
    v16bf b = load_b_qt(qrow, ks * 32);
    acc = wmma_bf16(a, b, acc);
  }
  const float scale = 0.125f; // 1/sqrt(64)
  if (n < NHEAD) {
#pragma unroll
    for (int r = 0; r < 8; ++r)
      scores[((size_t)n * NTOK + i) * NTOK + (j0 + mb + r)] = scale * acc[r];
  }
}

__global__ __launch_bounds__(256) void softmax_kernel(
    const float* __restrict__ scores, unsigned short* __restrict__ Pbf) {
  const size_t row = blockIdx.x;
  const float* s = scores + row * NTOK;
  unsigned short* p = Pbf + row * NTOK;
  const int tid = threadIdx.x;
  __shared__ float red[256];
  float v[4];
  float mx = -1e30f;
#pragma unroll
  for (int t = 0; t < 4; ++t) { v[t] = s[tid + t * 256]; mx = fmaxf(mx, v[t]); }
  red[tid] = mx; __syncthreads();
  for (int off = 128; off > 0; off >>= 1) {
    if (tid < off) red[tid] = fmaxf(red[tid], red[tid + off]);
    __syncthreads();
  }
  mx = red[0]; __syncthreads();
  float sum = 0.f;
#pragma unroll
  for (int t = 0; t < 4; ++t) { v[t] = __expf(v[t] - mx); sum += v[t]; }
  red[tid] = sum; __syncthreads();
  for (int off = 128; off > 0; off >>= 1) {
    if (tid < off) red[tid] += red[tid + off];
    __syncthreads();
  }
  const float inv = 1.f / red[0];
#pragma unroll
  for (int t = 0; t < 4; ++t) p[tid + t * 256] = f2bf(v[t] * inv);
}

// attn[i, h*64+d] = sum_j P[h,i,j] * V[j, h*64+d]   (per-head GEMM; V pre-packed)
__global__ __launch_bounds__(128) void pv_kernel(
    const unsigned short* __restrict__ Pbf, const unsigned short* __restrict__ Vp,
    float* __restrict__ attn) {
  const int i0 = blockIdx.x * 16;
  const int h  = blockIdx.y;
  const int wave = threadIdx.x >> 5;   // d-subtile
  const int lane = threadIdx.x & 31;
  const int ntile = h * 4 + wave;      // (h*64 + wave*16)/16 ; DMODEL/16 = 32 tiles per k-step
  v8f acc = {};
  const unsigned short* Ap = Pbf + ((size_t)h * NTOK + i0) * NTOK;
  for (int ks = 0; ks < (NTOK >> 5); ++ks) {
    v16bf a = load_a_frag(Ap + ks * 32, NTOK);
    v16bf b = load_b_packed(Vp, ks * 32 + ntile);
    acc = wmma_bf16(a, b, acc);
  }
  const int n  = lane & 15;
  const int mb = (lane & 16) ? 8 : 0;
  const int col = h * DK + wave * 16 + n;
#pragma unroll
  for (int r = 0; r < 8; ++r)
    attn[(size_t)(i0 + mb + r) * DMODEL + col] = acc[r];
}

// t[h,e] = sum_j P[h,i,j]*S[i,j,e]  then  attn[i,h,:] += t @ Wsv + bsv
// (linearity: Wsv projection commutes with the softmax-weighted sum; sum(p)=1 carries bsv)
__global__ __launch_bounds__(128) void structout_kernel(
    const float* __restrict__ structure, const unsigned short* __restrict__ Pbf,
    const unsigned short* __restrict__ Wsvp, const float* __restrict__ bsv,
    float* __restrict__ attn) {
  const int i    = blockIdx.x;
  const int wave = threadIdx.x >> 5;   // e-subtile, then d-subtile
  const int lane = threadIdx.x & 31;
  const int tid  = threadIdx.x;
  __shared__ __align__(16) unsigned short Sjt[64][40];  // TRANSPOSED: [e][j], pad 40
  __shared__ float Tsh[16][65];
  v8f acc = {};
  const unsigned short* Ap = Pbf + (size_t)i * NTOK;   // A(m=h, k=j), row stride NTOK*NTOK
  for (int ks = 0; ks < 32; ++ks) {
    const float* src = structure + ((size_t)i * NTOK + ks * 32) * DEDGE;
    for (int idx = tid; idx < 32 * 64; idx += 128) {
      int r = idx >> 6, c = idx & 63;          // r = j row (coalesced read), c = e
      Sjt[c][r] = f2bf(src[(size_t)r * DEDGE + c]);
    }
    if (ks + 1 < 32)
      __builtin_prefetch(structure + ((size_t)i * NTOK + (ks + 1) * 32) * DEDGE + tid * 16, 0, 1);
    __syncthreads();
    v16bf a = load_a_frag_masked(Ap + ks * 32, NTOK * NTOK, NHEAD);
    v16bf b = load_bT_frag(&Sjt[wave * 16][0], 40);
    acc = wmma_bf16(a, b, acc);
    __syncthreads();
  }
  const int n  = lane & 15;
  const int mb = (lane & 16) ? 8 : 0;
#pragma unroll
  for (int r = 0; r < 8; ++r) Tsh[mb + r][wave * 16 + n] = acc[r];
  __syncthreads();

  v8f acc2 = {};
#pragma unroll
  for (int ks = 0; ks < 2; ++ks) {
    v16bf a = load_a_frag_f32(&Tsh[0][ks * 32], 65);
    v16bf b = load_b_packed(Wsvp, ks * 4 + wave);
    acc2 = wmma_bf16(a, b, acc2);
  }
  const int d = wave * 16 + n;
#pragma unroll
  for (int r = 0; r < 8; ++r) {
    int hh = mb + r;
    if (hh < NHEAD)
      attn[(size_t)i * DMODEL + hh * DK + d] += acc2[r] + bsv[d];
  }
}

// ---------- host ----------

extern "C" void kernel_launch(void* const* d_in, const int* in_sizes, int n_in,
                              void* d_out, int out_size, void* d_ws, size_t ws_size,
                              hipStream_t stream) {
  const float* x         = (const float*)d_in[0];
  const float* structure = (const float*)d_in[1];
  const float* Wq  = (const float*)d_in[2];
  const float* bq  = (const float*)d_in[3];
  const float* Wk  = (const float*)d_in[4];
  const float* bk  = (const float*)d_in[5];
  const float* Wv  = (const float*)d_in[6];
  const float* bv  = (const float*)d_in[7];
  const float* Wo  = (const float*)d_in[8];
  const float* bo  = (const float*)d_in[9];
  const float* Wsk = (const float*)d_in[10];
  const float* bsk = (const float*)d_in[11];
  const float* Wsv = (const float*)d_in[12];
  const float* bsv = (const float*)d_in[13];
  float* out = (float*)d_out;

  char* ws = (char*)d_ws;
  size_t off = 0;
  auto alloc = [&](size_t bytes) -> void* {
    void* p = ws + off;
    off = (off + bytes + 255) & ~(size_t)255;
    return p;
  };

  unsigned short* Xbf    = (unsigned short*)alloc((size_t)NTOK * DMODEL * 2);
  unsigned short* Wqp    = (unsigned short*)alloc((size_t)DMODEL * DMODEL * 2);
  unsigned short* Wkp    = (unsigned short*)alloc((size_t)DMODEL * DMODEL * 2);
  unsigned short* Wvp    = (unsigned short*)alloc((size_t)DMODEL * DMODEL * 2);
  unsigned short* Wop    = (unsigned short*)alloc((size_t)DMODEL * DMODEL * 2);
  unsigned short* Wskp   = (unsigned short*)alloc((size_t)DEDGE * DK * 2);
  unsigned short* Wsvp   = (unsigned short*)alloc((size_t)DEDGE * DK * 2);
  unsigned short* Qbf    = (unsigned short*)alloc((size_t)NTOK * DMODEL * 2);
  unsigned short* Kbf    = (unsigned short*)alloc((size_t)NTOK * DMODEL * 2);
  unsigned short* Vp     = (unsigned short*)alloc((size_t)NTOK * DMODEL * 2);
  unsigned short* attnbf = (unsigned short*)alloc((size_t)NTOK * DMODEL * 2);
  float*          attnF  = (float*)alloc((size_t)NTOK * DMODEL * 4);
  float*          scores = (float*)alloc((size_t)NHEAD * NTOK * NTOK * 4);
  unsigned short* Pbf    = (unsigned short*)alloc((size_t)NHEAD * NTOK * NTOK * 2);
  (void)ws_size; (void)in_sizes; (void)n_in; (void)out_size;

  // bf16 conversion (A operand) + fragment-major packing (B operands)
  cvt_f32_bf16<<<512, 256, 0, stream>>>(x, Xbf, NTOK * DMODEL);
  pack_b_bf16<<<512, 256, 0, stream>>>(Wq, Wqp, DMODEL, DMODEL);
  pack_b_bf16<<<512, 256, 0, stream>>>(Wk, Wkp, DMODEL, DMODEL);
  pack_b_bf16<<<512, 256, 0, stream>>>(Wv, Wvp, DMODEL, DMODEL);
  pack_b_bf16<<<512, 256, 0, stream>>>(Wo, Wop, DMODEL, DMODEL);
  pack_b_bf16<<<16, 256, 0, stream>>>(Wsk, Wskp, DEDGE, DK);
  pack_b_bf16<<<16, 256, 0, stream>>>(Wsv, Wsvp, DEDGE, DK);

  // Q/K/V projections (V written directly in packed-B layout)
  dim3 gP(NTOK / 16, DMODEL / 128);
  gemm_bf16_kernel<<<gP, 256, 0, stream>>>(Xbf, Wqp, bq, nullptr, Qbf, NTOK, DMODEL, DMODEL, 1, 0);
  gemm_bf16_kernel<<<gP, 256, 0, stream>>>(Xbf, Wkp, bk, nullptr, Kbf, NTOK, DMODEL, DMODEL, 1, 0);
  gemm_bf16_kernel<<<gP, 256, 0, stream>>>(Xbf, Wvp, bv, nullptr, Vp,  NTOK, DMODEL, DMODEL, 0, 1);

  // fused QK + structure-key scores  (structure pass #1)
  scores_kernel<<<dim3(8, NTOK), 256, 0, stream>>>(structure, Qbf, Kbf, Wskp, bsk, scores);

  // softmax rows -> bf16 P
  softmax_kernel<<<NHEAD * NTOK, 256, 0, stream>>>(scores, Pbf);

  // value aggregation
  pv_kernel<<<dim3(NTOK / 16, NHEAD), 128, 0, stream>>>(Pbf, Vp, attnF);

  // structure-value aggregation via linearity trick (structure pass #2)
  structout_kernel<<<NTOK, 128, 0, stream>>>(structure, Pbf, Wsvp, bsv, attnF);

  // final projection -> d_out
  cvt_f32_bf16<<<512, 256, 0, stream>>>(attnF, attnbf, NTOK * DMODEL);
  gemm_bf16_kernel<<<gP, 256, 0, stream>>>(attnbf, Wop, bo, out, nullptr, NTOK, DMODEL, DMODEL, 0, 0);
}
